// LoRA_Attention_3186865734496
// MI455X (gfx1250) — matrix operand order repaired and checked by reading, not compiled, not run on hardware
//
#include <hip/hip_runtime.h>

#define DIMX   768
#define N_TOK  2304
#define HEADS  12
#define HD     64
#define GRID_HW 48
#define RANK   8

typedef __bf16 bf16_t;
typedef bf16_t v16bf __attribute__((ext_vector_type(16)));
typedef float  v8f   __attribute__((ext_vector_type(8)));
typedef unsigned short ushort_t;

__device__ __forceinline__ unsigned short f2bf(float f) {
    unsigned int u = __builtin_bit_cast(unsigned int, f);
    u += 0x7FFFu + ((u >> 16) & 1u);           // round-to-nearest-even
    return (unsigned short)(u >> 16);
}
__device__ __forceinline__ float bf2f(unsigned short u) {
    unsigned int v = ((unsigned int)u) << 16;
    return __builtin_bit_cast(float, v);
}

// Load 16 contiguous bf16 (32B) -> B-operand vector (slots ascending in memory).
__device__ __forceinline__ v16bf ld_op32(const ushort_t* p) {
    union { uint4 u[2]; v16bf v; } t;
    t.u[0] = *(const uint4*)(p);
    t.u[1] = *(const uint4*)(p + 8);
    return t.v;
}
// A-operand: slots 0..15 = row[k0+16*grp .. +15].
__device__ __forceinline__ v16bf ld_opA(const ushort_t* row, int grp) {
    union { uint4 u[2]; v16bf v; } t;
    t.u[0] = *(const uint4*)(row + 16 * grp);
    t.u[1] = *(const uint4*)(row + 16 * grp + 8);
    return t.v;
}

// ---------------------------------------------------------------------------
// Kernel 1: fold LoRA (W + A@B), convert to bf16, store TRANSPOSED (WT[n][k])
// so GEMM B-operands are per-lane contiguous.
// ---------------------------------------------------------------------------
__global__ void prep_weights_k(const float* __restrict__ Wq, const float* __restrict__ Wk,
                               const float* __restrict__ Wv, const float* __restrict__ Wp,
                               const float* __restrict__ Aq, const float* __restrict__ Bq,
                               const float* __restrict__ Ak, const float* __restrict__ Bk,
                               const float* __restrict__ Av, const float* __restrict__ Bv,
                               ushort_t* __restrict__ WQT, ushort_t* __restrict__ WKT,
                               ushort_t* __restrict__ WVT, ushort_t* __restrict__ WPT) {
    int t = blockIdx.x * blockDim.x + threadIdx.x;
    if (t >= DIMX * DIMX) return;
    int i = t % DIMX;        // input dim (contraction)
    int j = t / DIMX;        // output dim
    float wq = Wq[(size_t)i * DIMX + j];
    float wk = Wk[(size_t)i * DIMX + j];
    float wv = Wv[(size_t)i * DIMX + j];
#pragma unroll
    for (int r = 0; r < RANK; ++r) {
        wq += Aq[i * RANK + r] * Bq[r * DIMX + j];
        wk += Ak[i * RANK + r] * Bk[r * DIMX + j];
        wv += Av[i * RANK + r] * Bv[r * DIMX + j];
    }
    size_t o = (size_t)j * DIMX + i;   // transposed, coalesced store
    WQT[o] = f2bf(wq); WKT[o] = f2bf(wk); WVT[o] = f2bf(wv);
    WPT[o] = f2bf(Wp[(size_t)i * DIMX + j]);
}

__global__ void cvt_x_k(const float* __restrict__ x, ushort_t* __restrict__ XB) {
    int t = blockIdx.x * blockDim.x + threadIdx.x;
    if (t < N_TOK * DIMX) XB[t] = f2bf(x[t]);
}

// ---------------------------------------------------------------------------
// Kernel 2: bf16 WMMA GEMM  C[M x 768] = A[M x 768] @ BT^T + bias.
// One wave per 16x64 output tile; all operands loaded directly from global
// as per-lane-contiguous 16B/32B chunks (no LDS, no barriers).
// MODE 0: fp32 plain [tok][768]  (final projection -> d_out)
// MODE 1: bf16 head-split [hd][tok][64]   (Q, K)
// MODE 2: bf16 transposed  [hd][dim][tok] (V), packed b128 stores
// ---------------------------------------------------------------------------
template <int MODE>
__global__ __launch_bounds__(32) void wmma_gemm_k(
        const ushort_t* __restrict__ A, const ushort_t* __restrict__ BT,
        const float* __restrict__ bias, float* __restrict__ outF,
        ushort_t* __restrict__ outH) {
    const int m0 = blockIdx.x * 16;
    const int n0 = blockIdx.y * 64;
    const int lane = threadIdx.x;
    const int mrow = lane & 15, grp = lane >> 4;

    const ushort_t* arow = A + (size_t)(m0 + mrow) * DIMX;
    const ushort_t* brow[4];
#pragma unroll
    for (int t = 0; t < 4; ++t)
        brow[t] = BT + (size_t)(n0 + 16 * t + mrow) * DIMX + 16 * grp;

    v8f acc[4] = {v8f{}, v8f{}, v8f{}, v8f{}};
    for (int kk = 0; kk < DIMX; kk += 32) {
        __builtin_prefetch(arow + kk + 128, 0, 1);
        v16bf av = ld_opA(arow + kk, grp);
#pragma unroll
        for (int t = 0; t < 4; ++t) {
            v16bf bv = ld_op32(brow[t] + kk);
            acc[t] = __builtin_amdgcn_wmma_f32_16x16x32_bf16(false, av, false, bv,
                                                             (short)0, acc[t], false, false);
        }
    }
    // C layout: vgpr r -> row r + 8*grp, col = mrow (+16*t)
#pragma unroll
    for (int t = 0; t < 4; ++t) {
        int col = n0 + 16 * t + mrow;
        float bcol = bias[col];
        if (MODE == 2) {
            int hd = col >> 6, c = col & 63;
            union { ushort_t s[8]; uint4 u; } pk;
#pragma unroll
            for (int r = 0; r < 8; ++r) pk.s[r] = f2bf(acc[t][r] + bcol);
            *(uint4*)(outH + ((size_t)hd * HD + c) * N_TOK + m0 + 8 * grp) = pk.u;
        } else {
#pragma unroll
            for (int r = 0; r < 8; ++r) {
                int row = m0 + r + 8 * grp;
                float v = acc[t][r] + bcol;
                if (MODE == 0) {
                    outF[(size_t)row * DIMX + col] = v;
                } else {  // MODE 1
                    int hd = col >> 6, c = col & 63;
                    outH[((size_t)hd * N_TOK + row) * HD + c] = f2bf(v);
                }
            }
        }
    }
}

// ---------------------------------------------------------------------------
// Kernel 3: decomposed rel-pos bias tables.
// ---------------------------------------------------------------------------
__global__ void rel_bias_k(const ushort_t* __restrict__ QB,
                           const float* __restrict__ rph, const float* __restrict__ rpw,
                           float* __restrict__ RELH, float* __restrict__ RELW) {
    int t = blockIdx.x * blockDim.x + threadIdx.x;
    if (t >= HEADS * N_TOK * GRID_HW) return;
    int kk = t % GRID_HW;
    int q  = (t / GRID_HW) % N_TOK;
    int hd = t / (GRID_HW * N_TOK);
    int h = q / GRID_HW, w = q % GRID_HW;
    const ushort_t* qp = QB + ((size_t)hd * N_TOK + q) * HD;
    const float* rh = rph + (size_t)(h - kk + GRID_HW - 1) * HD;
    const float* rw = rpw + (size_t)(w - kk + GRID_HW - 1) * HD;
    float sh = 0.f, sw = 0.f;
#pragma unroll 8
    for (int c = 0; c < HD; ++c) {
        float qv = bf2f(qp[c]);
        sh += qv * rh[c];
        sw += qv * rw[c];
    }
    RELH[t] = sh;
    RELW[t] = sw;
}

// ---------------------------------------------------------------------------
// Kernel 4: fused flash attention. One wave = 16 queries of one head.
// Q/K/V operands loaded directly from global in WMMA lane layouts; only the
// P (probability) tile round-trips through 1KB of LDS for the C->A transform.
// ---------------------------------------------------------------------------
__global__ __launch_bounds__(32) void flash_attn_k(
        const ushort_t* __restrict__ QB, const ushort_t* __restrict__ KB,
        const ushort_t* __restrict__ VT,
        const float* __restrict__ RELH, const float* __restrict__ RELW,
        ushort_t* __restrict__ AO) {
    __shared__ ushort_t p_lds[16 * 32];

    const int q0 = blockIdx.x * 16;
    const int hd = blockIdx.y;
    const int lane = threadIdx.x;
    const int mrow = lane & 15, grp = lane >> 4;
    const float scale = 0.125f;   // 1/sqrt(64)

    const ushort_t* Kh = KB + (size_t)hd * N_TOK * HD;
    const ushort_t* Vh = VT + (size_t)hd * HD * N_TOK;
    const float* relh = RELH + (size_t)hd * N_TOK * GRID_HW;
    const float* relw = RELW + (size_t)hd * N_TOK * GRID_HW;

    // Q A-operands (16x64 tile, K split 0..31 / 32..63), direct from global.
    const ushort_t* qrow = QB + ((size_t)hd * N_TOK + q0 + mrow) * HD;
    v16bf aq0 = ld_opA(qrow, grp);
    v16bf aq1 = ld_opA(qrow + 32, grp);

    v8f o[4] = {v8f{}, v8f{}, v8f{}, v8f{}};
    float mi[8], li[8];
#pragma unroll
    for (int r = 0; r < 8; ++r) { mi[r] = -1e30f; li[r] = 0.f; }

    for (int m0 = 0; m0 < N_TOK; m0 += 32) {
        __builtin_prefetch(Kh + (size_t)(m0 + 32 + mrow) * HD, 0, 1);
        // ---- S = q @ k^T, two 16-key subtiles, K(head_dim)=64 ----
        v8f c[2];
#pragma unroll
        for (int s = 0; s < 2; ++s) {
            const ushort_t* krow = Kh + (size_t)(m0 + 16 * s + mrow) * HD + 16 * grp;
            v16bf b0 = ld_op32(krow);        // dims 0..31 chunk
            v16bf b1 = ld_op32(krow + 32);   // dims 32..63 chunk
            v8f t = {};
            t = __builtin_amdgcn_wmma_f32_16x16x32_bf16(false, aq0, false, b0,
                                                        (short)0, t, false, false);
            t = __builtin_amdgcn_wmma_f32_16x16x32_bf16(false, aq1, false, b1,
                                                        (short)0, t, false, false);
            c[s] = t;
        }

        // ---- scale + rel-pos bias + online softmax ----
        float p0[8], p1[8], lm[8];
#pragma unroll
        for (int r = 0; r < 8; ++r) {
            int q = q0 + r + 8 * grp;
            int key0 = m0 + mrow, key1 = m0 + 16 + mrow;
            float s0 = c[0][r] * scale
                     + relh[(size_t)q * GRID_HW + key0 / GRID_HW]
                     + relw[(size_t)q * GRID_HW + key0 % GRID_HW];
            float s1 = c[1][r] * scale
                     + relh[(size_t)q * GRID_HW + key1 / GRID_HW]
                     + relw[(size_t)q * GRID_HW + key1 % GRID_HW];
            p0[r] = s0; p1[r] = s1;
            lm[r] = fmaxf(s0, s1);
        }
#pragma unroll
        for (int off = 1; off < 16; off <<= 1)
#pragma unroll
            for (int r = 0; r < 8; ++r)
                lm[r] = fmaxf(lm[r], __shfl_xor(lm[r], off, 32));

        float alpha[8], rs[8];
#pragma unroll
        for (int r = 0; r < 8; ++r) {
            float mn = fmaxf(mi[r], lm[r]);
            alpha[r] = __expf(mi[r] - mn);
            mi[r] = mn;
            p0[r] = __expf(p0[r] - mn);
            p1[r] = __expf(p1[r] - mn);
            rs[r] = p0[r] + p1[r];
        }
#pragma unroll
        for (int off = 1; off < 16; off <<= 1)
#pragma unroll
            for (int r = 0; r < 8; ++r)
                rs[r] += __shfl_xor(rs[r], off, 32);
#pragma unroll
        for (int r = 0; r < 8; ++r) {
            li[r] = li[r] * alpha[r] + rs[r];
#pragma unroll
            for (int t2 = 0; t2 < 4; ++t2) o[t2][r] *= alpha[r];
        }

        // ---- P (16x32) C-layout -> LDS -> A-layout ----
        __syncthreads();   // protect previous iteration's p_lds reads
#pragma unroll
        for (int r = 0; r < 8; ++r) {
            int row = r + 8 * grp;
            p_lds[row * 32 + mrow]      = f2bf(p0[r]);
            p_lds[row * 32 + 16 + mrow] = f2bf(p1[r]);
        }
        __syncthreads();
        v16bf pa = ld_opA(&p_lds[mrow * 32], grp);

        // ---- O += P @ V  (V transposed: dim-major, keys contiguous) ----
#pragma unroll
        for (int t2 = 0; t2 < 4; ++t2) {
            const ushort_t* vrow = Vh + (size_t)(16 * t2 + mrow) * N_TOK + m0 + 16 * grp;
            v16bf bv = ld_op32(vrow);
            o[t2] = __builtin_amdgcn_wmma_f32_16x16x32_bf16(false, pa, false, bv,
                                                            (short)0, o[t2], false, false);
        }
    }

    // normalize and write bf16 attn output, token-major [tok][768]
#pragma unroll
    for (int r = 0; r < 8; ++r) {
        float inv = 1.f / li[r];
        int q = q0 + r + 8 * grp;
#pragma unroll
        for (int t2 = 0; t2 < 4; ++t2) {
            int dim = 16 * t2 + mrow;
            AO[(size_t)q * DIMX + hd * HD + dim] = f2bf(o[t2][r] * inv);
        }
    }
}

// ---------------------------------------------------------------------------
// Host orchestration
// ---------------------------------------------------------------------------
extern "C" void kernel_launch(void* const* d_in, const int* in_sizes, int n_in,
                              void* d_out, int out_size, void* d_ws, size_t ws_size,
                              hipStream_t stream) {
    (void)in_sizes; (void)n_in; (void)out_size; (void)ws_size;
    const float* x   = (const float*)d_in[0];
    const float* Wq  = (const float*)d_in[1];
    const float* bq  = (const float*)d_in[2];
    const float* Wk  = (const float*)d_in[3];
    const float* bk  = (const float*)d_in[4];
    const float* Wv  = (const float*)d_in[5];
    const float* bv  = (const float*)d_in[6];
    const float* Wp  = (const float*)d_in[7];
    const float* bp  = (const float*)d_in[8];
    const float* rph = (const float*)d_in[9];
    const float* rpw = (const float*)d_in[10];
    const float* Aq  = (const float*)d_in[11];
    const float* Bq  = (const float*)d_in[12];
    const float* Ak  = (const float*)d_in[13];
    const float* Bk  = (const float*)d_in[14];
    const float* Av  = (const float*)d_in[15];
    const float* Bv  = (const float*)d_in[16];
    float* out = (float*)d_out;

    char* ws = (char*)d_ws;
    size_t off = 0;
    auto take = [&](size_t b) { size_t o = off; off += (b + 255) & ~(size_t)255; return o; };
    ushort_t* XB   = (ushort_t*)(ws + take((size_t)N_TOK * DIMX * 2));
    ushort_t* WQT  = (ushort_t*)(ws + take((size_t)DIMX * DIMX * 2));
    ushort_t* WKT  = (ushort_t*)(ws + take((size_t)DIMX * DIMX * 2));
    ushort_t* WVT  = (ushort_t*)(ws + take((size_t)DIMX * DIMX * 2));
    ushort_t* WPT  = (ushort_t*)(ws + take((size_t)DIMX * DIMX * 2));
    ushort_t* QB   = (ushort_t*)(ws + take((size_t)N_TOK * DIMX * 2));
    ushort_t* KB   = (ushort_t*)(ws + take((size_t)N_TOK * DIMX * 2));
    ushort_t* VT   = (ushort_t*)(ws + take((size_t)N_TOK * DIMX * 2));
    float*    RELH = (float*)(ws + take((size_t)HEADS * N_TOK * GRID_HW * 4));
    float*    RELW = (float*)(ws + take((size_t)HEADS * N_TOK * GRID_HW * 4));
    ushort_t* AO   = (ushort_t*)(ws + take((size_t)N_TOK * DIMX * 2));

    prep_weights_k<<<(DIMX * DIMX + 255) / 256, 256, 0, stream>>>(
        Wq, Wk, Wv, Wp, Aq, Bq, Ak, Bk, Av, Bv, WQT, WKT, WVT, WPT);
    cvt_x_k<<<(N_TOK * DIMX + 255) / 256, 256, 0, stream>>>(x, XB);

    dim3 ggrid(N_TOK / 16, DIMX / 64);
    wmma_gemm_k<1><<<ggrid, 32, 0, stream>>>(XB, WQT, bq, nullptr, QB);
    wmma_gemm_k<1><<<ggrid, 32, 0, stream>>>(XB, WKT, bk, nullptr, KB);
    wmma_gemm_k<2><<<ggrid, 32, 0, stream>>>(XB, WVT, bv, nullptr, VT);

    rel_bias_k<<<(HEADS * N_TOK * GRID_HW + 255) / 256, 256, 0, stream>>>(
        QB, rph, rpw, RELH, RELW);

    flash_attn_k<<<dim3(N_TOK / 16, HEADS), 32, 0, stream>>>(
        QB, KB, VT, RELH, RELW, AO);

    wmma_gemm_k<0><<<ggrid, 32, 0, stream>>>(AO, WPT, bp, out, nullptr);
}
